// PLEModel_38792144617891
// MI455X (gfx1250) — compile-verified, matrix operand order checked by reading, not run on hardware
//
#include <hip/hip_runtime.h>
#include <hip/hip_bf16.h>

// ---------------- problem constants ----------------
#define BATCH   8192
#define NF      20
#define NV      1000
#define NE      32
#define NDD     10
#define IN0D    960      // (F+ND)*E
#define NT      2
#define NK      4
#define NS      4
#define DDIM    512
#define HDIM    256
#define GDIM    128      // H/2
#define NEXP    6144     // (T*K + S) * D
#define TKD     4096     // T*K*D
#define NGATE   8        // K+S
#define TEMP_INV 0.5f

typedef __attribute__((ext_vector_type(16))) __bf16 v16bf;
typedef __attribute__((ext_vector_type(8)))  float  v8f;
typedef __attribute__((ext_vector_type(4)))  unsigned int uint32x4;
typedef __attribute__((ext_vector_type(8)))  int          int32x8;
typedef __attribute__((ext_vector_type(4)))  int          int32x4;

union FragBF {
  v16bf v;
  uint4 q[2];
};

__device__ __forceinline__ unsigned short f2bf(float f) {
  unsigned int u = __float_as_uint(f);
  u += 0x7FFFu + ((u >> 16) & 1u);         // round-to-nearest-even
  return (unsigned short)(u >> 16);
}
__device__ __forceinline__ float bf2f(unsigned short h) {
  return __uint_as_float(((unsigned int)h) << 16);
}

#if defined(__gfx1250__) && __has_builtin(__builtin_amdgcn_tensor_load_to_lds)
#define USE_TDM 1
#else
#define USE_TDM 0
#endif

#define LSTR 40   // padded LDS row stride in halfs (64B data + 16B pad)

#if USE_TDM
// Issue one TDM 2D tile load: 128 rows x 32 bf16 elements, global row stride
// ld_elems (in elements), into LDS at byte offset lds_off with +16B padding
// after every 64B row (matches LSTR). D# layout per CDNA5 ISA ch.8.
// NOTE: this toolchain exposes the 6-arg builtin form:
//   (uint32x4 g0, int32x8 g1, int32x4 g2, int32x4 g3, int32x8 g4, i32 cpol)
__device__ __forceinline__ void tdm_load_tile(unsigned long long gaddr,
                                              unsigned lds_off,
                                              unsigned ld_elems) {
  uint32x4 g0;
  g0[0] = 1u;                                        // count=1, user mode
  g0[1] = lds_off;                                   // lds_addr (bytes)
  g0[2] = (unsigned)(gaddr & 0xFFFFFFFFull);         // global_addr[31:0]
  g0[3] = (unsigned)((gaddr >> 32) & 0x01FFFFFFull)  // global_addr[56:32]
          | (2u << 30);                              // type=2 ("image")
  int32x8 g1;
  // data_size=1 (2B) | pad_enable | pad_interval=3 (16 DW) | pad_amount=3 (4 DW)
  g1[0] = (int)((1u << 16) | (1u << 20) | (3u << 22) | (3u << 25));
  // atomic_barrier_addr=0; tensor_dim0 = 2^30 (no OOB); tensor_dim1 = 2^30
  g1[1] = (int)((0x40000000u & 0xFFFFu) << 16);              // dim0 lo16 = 0
  g1[2] = (int)((0x40000000u >> 16) | ((0x40000000u & 0xFFFFu) << 16)); // 0x4000
  g1[3] = (int)((0x40000000u >> 16) | (32u << 16));          // dim1 hi16 | tile_dim0=32
  g1[4] = (int)(128u);                                       // tile_dim1=128, tile_dim2=0
  g1[5] = (int)ld_elems;                                     // tensor_dim0_stride lo32
  g1[6] = 0;                                                 // stride0 hi16 | stride1 lo16
  g1[7] = 0;                                                 // stride1 hi32
  int32x4 z4 = {0, 0, 0, 0};
  int32x8 z8 = {0, 0, 0, 0, 0, 0, 0, 0};
  __builtin_amdgcn_tensor_load_to_lds(g0, g1, z4, z4, z8, 0);
}
#endif

// ---------------------------------------------------------------------------
// Tiled WMMA GEMM:  C(bf16) = epilogue( A(bf16, MxK) * Bt(bf16, NxK) + bias )
// B is PRE-TRANSPOSED (N x K, row stride ldb). Block tile 128x128, K-step 32,
// 8 waves (4x2), wave tile 32x64 = 2x4 WMMAs.
// mode 0: relu(acc + bias)        mode 1: relu(BN(acc + bias))
// ---------------------------------------------------------------------------
__global__ __launch_bounds__(256, 2) void gemm_bf16_wmma(
    const unsigned short* __restrict__ A, int lda,
    const unsigned short* __restrict__ Bt, int ldb,
    unsigned short* __restrict__ C, int ldc,
    const float* __restrict__ bias,
    const float* __restrict__ bng, const float* __restrict__ bnb,
    const float* __restrict__ bnm, const float* __restrict__ bnv,
    int K, int mode) {
  const int tid   = threadIdx.x;
  const int lane  = tid & 31;
  const int wid   = tid >> 5;
  const int waveM = wid >> 1;          // 0..3
  const int waveN = wid & 1;           // 0..1
  const int fr    = lane & 15;         // row/col within a 16x16 tile
  const int kb    = (lane >> 4) * 8;   // K base inside 32-chunk (ISA layout)

  __shared__ unsigned short As[2][128 * LSTR];
  __shared__ unsigned short Bs[2][128 * LSTR];   // [n][k]

  v8f acc[2][4];
#pragma unroll
  for (int i = 0; i < 2; ++i)
#pragma unroll
    for (int j = 0; j < 4; ++j)
#pragma unroll
      for (int r = 0; r < 8; ++r) acc[i][j][r] = 0.0f;

  const unsigned short* Ag = A + (size_t)(blockIdx.y * 128) * lda;
  const unsigned short* Bg = Bt + (size_t)(blockIdx.x * 128) * ldb;
  const int Kt = K >> 5;

#if USE_TDM
  const bool issuer = (wid == 0);                 // wave-uniform: one wave DMAs
  const unsigned long long Aga = (unsigned long long)(uintptr_t)Ag;
  const unsigned long long Bga = (unsigned long long)(uintptr_t)Bg;
  const unsigned ldsA0 = (unsigned)(uintptr_t)&As[0][0];
  const unsigned ldsA1 = (unsigned)(uintptr_t)&As[1][0];
  const unsigned ldsB0 = (unsigned)(uintptr_t)&Bs[0][0];
  const unsigned ldsB1 = (unsigned)(uintptr_t)&Bs[1][0];
  if (issuer) {
    tdm_load_tile(Aga, ldsA0, (unsigned)lda);
    tdm_load_tile(Bga, ldsB0, (unsigned)ldb);
    __builtin_amdgcn_s_wait_tensorcnt(0);
  }
  __syncthreads();
#else
  auto stageA = [&](int buf, int k0) {
    int row = tid >> 1, seg = tid & 1;                 // 16 halfs per thread
    const unsigned short* src = Ag + (size_t)row * lda + k0 + seg * 16;
    uint4 p0 = *(const uint4*)(src);
    uint4 p1 = *(const uint4*)(src + 8);
    *(uint4*)&As[buf][row * LSTR + seg * 16]     = p0;
    *(uint4*)&As[buf][row * LSTR + seg * 16 + 8] = p1;
  };
  auto stageB = [&](int buf, int k0) {
    int row = tid >> 1, seg = tid & 1;
    const unsigned short* src = Bg + (size_t)row * ldb + k0 + seg * 16;
    uint4 p0 = *(const uint4*)(src);
    uint4 p1 = *(const uint4*)(src + 8);
    *(uint4*)&Bs[buf][row * LSTR + seg * 16]     = p0;
    *(uint4*)&Bs[buf][row * LSTR + seg * 16 + 8] = p1;
  };
  stageA(0, 0);
  stageB(0, 0);
  __syncthreads();
#endif

  for (int kt = 0; kt < Kt; ++kt) {
    const int cur = kt & 1;
#if USE_TDM
    if (kt + 1 < Kt && issuer) {       // prefetch next K-slab via TDM DMA
      const unsigned long long koff = (unsigned long long)(kt + 1) * 64ull; // 32 elems * 2B
      tdm_load_tile(Aga + koff, cur ? ldsA0 : ldsA1, (unsigned)lda);
      tdm_load_tile(Bga + koff, cur ? ldsB0 : ldsB1, (unsigned)ldb);
    }
#else
    if (kt + 1 < Kt) {
      stageA(cur ^ 1, (kt + 1) * 32);
      stageB(cur ^ 1, (kt + 1) * 32);
    }
#endif
    const unsigned short* Ab = &As[cur][(waveM * 32 + fr) * LSTR + kb];
    FragBF a0, a1;
    a0.q[0] = *(const uint4*)(Ab);
    a0.q[1] = *(const uint4*)(Ab + 16);
    a1.q[0] = *(const uint4*)(Ab + 16 * LSTR);
    a1.q[1] = *(const uint4*)(Ab + 16 * LSTR + 16);
    const unsigned short* Bb = &Bs[cur][(waveN * 64 + fr) * LSTR + kb];
#pragma unroll
    for (int j = 0; j < 4; ++j) {
      FragBF bfrag;
      bfrag.q[0] = *(const uint4*)(Bb + j * 16 * LSTR);
      bfrag.q[1] = *(const uint4*)(Bb + j * 16 * LSTR + 16);
      acc[0][j] = __builtin_amdgcn_wmma_f32_16x16x32_bf16(
          false, a0.v, false, bfrag.v, (short)0, acc[0][j], false, false);
      acc[1][j] = __builtin_amdgcn_wmma_f32_16x16x32_bf16(
          false, a1.v, false, bfrag.v, (short)0, acc[1][j], false, false);
    }
#if USE_TDM
    if (issuer) __builtin_amdgcn_s_wait_tensorcnt(0);
#endif
    __syncthreads();
  }

  // Epilogue. C/D layout: VGPR r -> row (base + r + (lane>=16 ? 8:0)), col = base + (lane&15)
  const int rowHalf = (lane >> 4) * 8;
#pragma unroll
  for (int i = 0; i < 2; ++i) {
#pragma unroll
    for (int j = 0; j < 4; ++j) {
      const int col = blockIdx.x * 128 + waveN * 64 + j * 16 + fr;
      const float bsv = bias[col];
      float sc = 1.0f, sh = 0.0f;
      if (mode == 1) {
        const float inv = rsqrtf(bnv[col] + 1e-5f);
        sc = bng[col] * inv;
        sh = bnb[col] - bnm[col] * sc;
      }
#pragma unroll
      for (int r = 0; r < 8; ++r) {
        const int row = blockIdx.y * 128 + waveM * 32 + i * 16 + rowHalf + r;
        float v = acc[i][j][r] + bsv;
        v = v * sc + sh;
        v = fmaxf(v, 0.0f);
        C[(size_t)row * ldc + col] = f2bf(v);
      }
    }
  }
}

// ---------------------------------------------------------------------------
// Embedding gather + small dense projection -> x0 (B x 960, bf16)
// ---------------------------------------------------------------------------
__global__ void embed_kernel(const int* __restrict__ sparse,
                             const float* __restrict__ dense,
                             const float* __restrict__ emb,
                             const float* __restrict__ dW,
                             const float* __restrict__ db,
                             unsigned short* __restrict__ x0) {
  const int idx = blockIdx.x * blockDim.x + threadIdx.x;
  if (idx >= BATCH * IN0D) return;
  const int b = idx / IN0D, c = idx - b * IN0D;
  float v;
  if (c < NF * NE) {
    const int f = c >> 5, e = c & 31;
    const int row = sparse[b * NF + f];
    v = emb[((size_t)f * NV + row) * NE + e];
  } else {
    const int j = c - NF * NE;     // 0..319
    v = db[j];
#pragma unroll
    for (int i = 0; i < NDD; ++i) v += dense[b * NDD + i] * dW[i * (NDD * NE) + j];
  }
  x0[idx] = f2bf(v);
}

// Pack task+shared expert weights into one TRANSPOSED bf16 (6144 x In) matrix.
__global__ void pack_expert_w(const float* __restrict__ Wt,   // (T*K, In, D)
                              const float* __restrict__ Wsh,  // (S, In, D)
                              unsigned short* __restrict__ out, int In) {
  const size_t idx = (size_t)blockIdx.x * blockDim.x + threadIdx.x;
  if (idx >= (size_t)In * NEXP) return;
  const int i = (int)(idx / NEXP), c = (int)(idx - (size_t)i * NEXP);
  float v;
  if (c < TKD) {
    const int ek = c / DDIM, d = c - ek * DDIM;
    v = Wt[((size_t)ek * In + i) * DDIM + d];
  } else {
    const int cs = c - TKD;
    const int s = cs / DDIM, d = cs - s * DDIM;
    v = Wsh[((size_t)s * In + i) * DDIM + d];
  }
  out[(size_t)c * In + i] = f2bf(v);      // transposed: row = output col
}

__global__ void pack_expert_b(const float* __restrict__ bt,
                              const float* __restrict__ bsh,
                              float* __restrict__ out) {
  const int c = blockIdx.x * blockDim.x + threadIdx.x;
  if (c >= NEXP) return;
  out[c] = (c < TKD) ? bt[c] : bsh[c - TKD];
}

// Transpose-convert: in (R x C, f32) -> out (C x R, bf16)
__global__ void cvt_bf16_t(const float* __restrict__ in,
                           unsigned short* __restrict__ out, int R, int C) {
  const int idx = blockIdx.x * blockDim.x + threadIdx.x;
  if (idx >= R * C) return;
  const int r = idx / C, c = idx - r * C;
  out[(size_t)c * R + r] = f2bf(in[idx]);
}

// ---------------------------------------------------------------------------
// Gate: one wave per (b, t). logits[e] = x . Wg[t,:,e] + bg; softmax(logit/T)
// ---------------------------------------------------------------------------
__global__ void gate_kernel(const unsigned short* __restrict__ x, int In,
                            const float* __restrict__ Wg,   // (T, In, 8)
                            const float* __restrict__ bg,   // (T, 8)
                            float* __restrict__ gw) {       // (B*T, 8)
  const int lane = threadIdx.x & 31;
  const int gid  = blockIdx.x * (blockDim.x >> 5) + (threadIdx.x >> 5);
  const int b = gid >> 1, t = gid & 1;
  float acc[NGATE];
#pragma unroll
  for (int e = 0; e < NGATE; ++e) acc[e] = 0.0f;
  for (int i = lane; i < In; i += 32) {
    const float xv = bf2f(x[(size_t)b * In + i]);
    const float* w = Wg + ((size_t)t * In + i) * NGATE;
#pragma unroll
    for (int e = 0; e < NGATE; ++e) acc[e] += xv * w[e];
  }
#pragma unroll
  for (int off = 16; off > 0; off >>= 1)
#pragma unroll
    for (int e = 0; e < NGATE; ++e) acc[e] += __shfl_xor(acc[e], off, 32);
  if (lane == 0) {
    float mx = -1e30f;
#pragma unroll
    for (int e = 0; e < NGATE; ++e) {
      acc[e] = (acc[e] + bg[t * NGATE + e]) * TEMP_INV;
      mx = fmaxf(mx, acc[e]);
    }
    float den = 0.0f;
#pragma unroll
    for (int e = 0; e < NGATE; ++e) { acc[e] = __expf(acc[e] - mx); den += acc[e]; }
    const float rd = 1.0f / den;
#pragma unroll
    for (int e = 0; e < NGATE; ++e) gw[(size_t)gid * NGATE + e] = acc[e] * rd;
  }
}

// Mix experts with gates; emit tout (B,T,D bf16) and next input (B,D bf16).
__global__ void mix_kernel(const unsigned short* __restrict__ EO,  // (B, 6144)
                           const float* __restrict__ gw,           // (B*T, 8)
                           unsigned short* __restrict__ tout,      // (B, T*D)
                           unsigned short* __restrict__ xn) {      // (B, D)
  const int idx = blockIdx.x * blockDim.x + threadIdx.x;
  if (idx >= BATCH * DDIM) return;
  const int b = idx / DDIM, d = idx - b * DDIM;
  const unsigned short* eo = EO + (size_t)b * NEXP;
  float xs = 0.0f;
#pragma unroll
  for (int t = 0; t < NT; ++t) {
    const float* g = gw + ((size_t)b * NT + t) * NGATE;
    float a = 0.0f;
#pragma unroll
    for (int e = 0; e < NK; ++e) a += g[e] * bf2f(eo[(t * NK + e) * DDIM + d]);
#pragma unroll
    for (int s = 0; s < NS; ++s) a += g[NK + s] * bf2f(eo[TKD + s * DDIM + d]);
    tout[((size_t)b * NT + t) * DDIM + d] = f2bf(a);
    xs += a;
  }
  xn[(size_t)b * DDIM + d] = f2bf(xs * 0.5f);
}

// Final per-task dot + sigmoid heads -> out (3, B) float
__global__ void final_kernel(const unsigned short* __restrict__ h2, // (B, T*128)
                             const float* __restrict__ tw3,         // (T, 128)
                             const float* __restrict__ tb3,         // (T,)
                             float* __restrict__ out) {
  const int b = blockIdx.x * blockDim.x + threadIdx.x;
  if (b >= BATCH) return;
  float logit[NT];
#pragma unroll
  for (int t = 0; t < NT; ++t) {
    float a = tb3[t];
    const unsigned short* h = h2 + ((size_t)b * NT + t) * GDIM;
#pragma unroll 8
    for (int g = 0; g < GDIM; ++g) a += bf2f(h[g]) * tw3[t * GDIM + g];
    logit[t] = a;
  }
  const float ctr = 1.0f / (1.0f + __expf(-logit[0]));
  const float cvr = 1.0f / (1.0f + __expf(-logit[1]));
  const float ctcvr = ctr * cvr;
  out[b] = ctr;
  out[BATCH + b] = ctcvr;
  out[2 * BATCH + b] = ctcvr;
}

// ---------------------------------------------------------------------------
extern "C" void kernel_launch(void* const* d_in, const int* in_sizes, int n_in,
                              void* d_out, int out_size, void* d_ws, size_t ws_size,
                              hipStream_t stream) {
  const int*   sparse = (const int*)d_in[0];
  const float* dense  = (const float*)d_in[1];
  const float* emb    = (const float*)d_in[2];
  const float* dW     = (const float*)d_in[3];
  const float* db     = (const float*)d_in[4];
  const float *Wt[3], *bt[3], *Ws[3], *bs[3], *Wg[3], *bg[3];
  for (int l = 0; l < 3; ++l) {
    const int base = 5 + l * 6;
    Wt[l] = (const float*)d_in[base + 0];
    bt[l] = (const float*)d_in[base + 1];
    Ws[l] = (const float*)d_in[base + 2];
    bs[l] = (const float*)d_in[base + 3];
    Wg[l] = (const float*)d_in[base + 4];
    bg[l] = (const float*)d_in[base + 5];
  }
  const float* tw1 = (const float*)d_in[23];
  const float* tb1 = (const float*)d_in[24];
  const float* bng = (const float*)d_in[25];
  const float* bnb = (const float*)d_in[26];
  const float* bnm = (const float*)d_in[27];
  const float* bnv = (const float*)d_in[28];
  const float* tw2 = (const float*)d_in[29];
  const float* tb2 = (const float*)d_in[30];
  const float* tw3 = (const float*)d_in[31];
  const float* tb3 = (const float*)d_in[32];
  float* out = (float*)d_out;

  // ---- workspace carve-out (256B aligned) ----
  char* p = (char*)d_ws;
  auto carve = [&](size_t bytes) {
    char* r = p;
    p += (bytes + 255) & ~(size_t)255;
    return r;
  };
  unsigned short* wcat[3];                       // transposed: (6144 x In)
  wcat[0] = (unsigned short*)carve((size_t)NEXP * IN0D * 2);
  wcat[1] = (unsigned short*)carve((size_t)NEXP * DDIM * 2);
  wcat[2] = (unsigned short*)carve((size_t)NEXP * DDIM * 2);
  float* bcat[3];
  for (int l = 0; l < 3; ++l) bcat[l] = (float*)carve(NEXP * 4);
  unsigned short* tw1b = (unsigned short*)carve((size_t)NT * HDIM * DDIM * 2); // (T,H,D)
  unsigned short* tw2b = (unsigned short*)carve((size_t)NT * GDIM * HDIM * 2); // (T,G,H)
  unsigned short* x0   = (unsigned short*)carve((size_t)BATCH * IN0D * 2);
  unsigned short* xn   = (unsigned short*)carve((size_t)BATCH * DDIM * 2);
  unsigned short* EO   = (unsigned short*)carve((size_t)BATCH * NEXP * 2);
  float*          gwb  = (float*)carve((size_t)BATCH * NT * NGATE * 4);
  unsigned short* toutb= (unsigned short*)carve((size_t)BATCH * NT * DDIM * 2);
  unsigned short* h1   = (unsigned short*)carve((size_t)BATCH * NT * HDIM * 2);
  unsigned short* h2   = (unsigned short*)carve((size_t)BATCH * NT * GDIM * 2);

  // ---- pack / convert weights (per-call; launch is stateless) ----
  {
    size_t n0 = (size_t)IN0D * NEXP;
    pack_expert_w<<<(unsigned)((n0 + 255) / 256), 256, 0, stream>>>(Wt[0], Ws[0], wcat[0], IN0D);
    size_t n1 = (size_t)DDIM * NEXP;
    pack_expert_w<<<(unsigned)((n1 + 255) / 256), 256, 0, stream>>>(Wt[1], Ws[1], wcat[1], DDIM);
    pack_expert_w<<<(unsigned)((n1 + 255) / 256), 256, 0, stream>>>(Wt[2], Ws[2], wcat[2], DDIM);
    for (int l = 0; l < 3; ++l)
      pack_expert_b<<<NEXP / 256, 256, 0, stream>>>(bt[l], bs[l], bcat[l]);
    for (int t = 0; t < NT; ++t) {
      cvt_bf16_t<<<(DDIM * HDIM) / 256, 256, 0, stream>>>(
          tw1 + (size_t)t * DDIM * HDIM, tw1b + (size_t)t * HDIM * DDIM, DDIM, HDIM);
      cvt_bf16_t<<<(HDIM * GDIM) / 256, 256, 0, stream>>>(
          tw2 + (size_t)t * HDIM * GDIM, tw2b + (size_t)t * GDIM * HDIM, HDIM, GDIM);
    }
  }

  // ---- embedding + dense -> x0 ----
  embed_kernel<<<(BATCH * IN0D) / 256, 256, 0, stream>>>(sparse, dense, emb, dW, db, x0);

  // ---- three PLE layers ----
  const unsigned short* xin = x0;
  int In = IN0D;
  for (int l = 0; l < 3; ++l) {
    gate_kernel<<<(BATCH * NT) / 8, 256, 0, stream>>>(xin, In, Wg[l], bg[l], gwb);
    gemm_bf16_wmma<<<dim3(NEXP / 128, BATCH / 128), 256, 0, stream>>>(
        xin, In, wcat[l], In, EO, NEXP, bcat[l],
        nullptr, nullptr, nullptr, nullptr, In, /*mode=*/0);
    mix_kernel<<<(BATCH * DDIM) / 256, 256, 0, stream>>>(EO, gwb, toutb, xn);
    xin = xn;
    In = DDIM;
  }

  // ---- per-task towers ----
  for (int t = 0; t < NT; ++t) {
    // h1 = relu(BN(tout_t @ tw1_t + tb1_t))   (8192x512)*(512x256)
    gemm_bf16_wmma<<<dim3(HDIM / 128, BATCH / 128), 256, 0, stream>>>(
        toutb + t * DDIM, NT * DDIM,
        tw1b + (size_t)t * HDIM * DDIM, DDIM,
        h1 + t * HDIM, NT * HDIM,
        tb1 + t * HDIM, bng + t * HDIM, bnb + t * HDIM, bnm + t * HDIM, bnv + t * HDIM,
        DDIM, /*mode=*/1);
    // h2 = relu(h1_t @ tw2_t + tb2_t)         (8192x256)*(256x128)
    gemm_bf16_wmma<<<dim3(GDIM / 128, BATCH / 128), 256, 0, stream>>>(
        h1 + t * HDIM, NT * HDIM,
        tw2b + (size_t)t * GDIM * HDIM, HDIM,
        h2 + t * GDIM, NT * GDIM,
        tb2 + t * GDIM, nullptr, nullptr, nullptr, nullptr,
        HDIM, /*mode=*/0);
  }

  // ---- heads ----
  final_kernel<<<BATCH / 256, 256, 0, stream>>>(h2, tw3, tb3, out);
  (void)in_sizes; (void)n_in; (void)out_size; (void)ws_size;
}
